// LSTM_CRF_Model_88003879895338
// MI455X (gfx1250) — matrix operand order repaired and checked by reading, not compiled, not run on hardware
//
#include <hip/hip_runtime.h>
#include <hip/hip_bf16.h>

// ---- problem constants (must match reference) ----
#define B_      64
#define T_      512
#define EMB_    128
#define RADE_   64
#define DIN_    192     // EMB + RAD_EMB
#define HID_    512
#define G4_     2048    // 4*HID
#define CLS_    10
#define START_ID_ 8
#define NEG_    (-10000.0f)

typedef __attribute__((ext_vector_type(16))) _Float16 v16h;
typedef __attribute__((ext_vector_type(8)))  _Float16 v8h;
typedef __attribute__((ext_vector_type(8)))  float    v8f;

// -------------------------------------------------------------------------
// WMMA helper: D = A(16x32 f16) * B(32x16 f16) + C(f32)
// -------------------------------------------------------------------------
__device__ __forceinline__ v8f wmma_f16(v16h a, v16h b, v8f c) {
    return __builtin_amdgcn_wmma_f32_16x16x32_f16(
        /*neg_a=*/false, a, /*neg_b=*/false, b,
        /*c_mod=*/(short)0, c, /*reuse_a=*/false, /*reuse_b=*/false);
}

// Load a 16-half A fragment for one lane from a row-major global source.
// CDNA5 A layout: lanes 0-15: K k0..k0+7 , k0+16..k0+23 (k0 = kg*32)
//                 lanes 16-31: K k0+8..k0+15, k0+24..k0+31
__device__ __forceinline__ v16h load_afrag(const _Float16* __restrict__ base,
                                           int row, int rowStride, int k0) {
    const _Float16* p = base + (size_t)row * rowStride + k0;
    v8h lo = *(const v8h*)(p);        // 16 bytes
    v8h hi = *(const v8h*)(p + 16);   // next K-group half
    return __builtin_shufflevector(lo, hi, 0,1,2,3,4,5,6,7,8,9,10,11,12,13,14,15);
}

// -------------------------------------------------------------------------
// Pack f32 weight W[N=2048][K] into f16 B-panel:
//   panel[(kg*(G4/16)+nt)*512 + lane*16 + s]   (32 contiguous bytes per lane)
//   lane = (n%16) + 16*(((k%32)/8)%2),  s = (k%8) + 8*((k%32)/16)
// -------------------------------------------------------------------------
__global__ void pack_weight_kernel(const float* __restrict__ W,
                                   _Float16* __restrict__ P, int K) {
    int total = G4_ * K;
    for (int e = blockIdx.x * blockDim.x + threadIdx.x; e < total;
         e += gridDim.x * blockDim.x) {
        int n = e / K, k = e % K;
        int nt = n >> 4, nl = n & 15;
        int kg = k >> 5, kk = k & 31;
        int lane = nl + (((kk >> 3) & 1) << 4);
        int s = (kk & 7) + ((kk >> 4) << 3);
        P[((size_t)(kg * (G4_ / 16) + nt)) * 512 + lane * 16 + s] = (_Float16)W[e];
    }
}

// -------------------------------------------------------------------------
// Embedding gather + f32->f16:  X16[t][b][d]
// -------------------------------------------------------------------------
__global__ void embed_kernel(const int* __restrict__ bd, const int* __restrict__ br,
                             const float* __restrict__ ce, const float* __restrict__ re,
                             _Float16* __restrict__ X) {
    int total = T_ * B_ * DIN_;
    for (int e = blockIdx.x * blockDim.x + threadIdx.x; e < total;
         e += gridDim.x * blockDim.x) {
        int d = e % DIN_;
        int tb = e / DIN_;
        int b = tb % B_;
        int t = tb / B_;
        float v;
        if (d < EMB_) { int id = bd[b * T_ + t]; v = ce[(size_t)id * EMB_ + d]; }
        else          { int id = br[b * T_ + t]; v = re[(size_t)id * RADE_ + (d - EMB_)]; }
        X[e] = (_Float16)v;
    }
}

__global__ void init_state_kernel(_Float16* hF, _Float16* hB, float* cF, float* cB) {
    int i = blockIdx.x * blockDim.x + threadIdx.x;
    if (i < B_ * HID_) {
        hF[i] = (_Float16)0.0f; hB[i] = (_Float16)0.0f;
        cF[i] = 0.0f;           cB[i] = 0.0f;
    }
}

// -------------------------------------------------------------------------
// One LSTM time step:  g = x_t @ Wih^T + h @ Whh^T + b ; gates ; update c,h.
//
// Grid: 64 WGs x 256 threads (8 waves).
//   wg>>1  -> 16-col group of hidden units (j0 = (wg>>1)*16)
//   wg&1   -> batch-row half (rows rbase..rbase+31)
//   wave wv: gate = wv>>1 (i,f,g,o), local M-tile mtl = wv&1
// Each wave computes ONE 16x16 D tile over K=704 (22 WMMA, 2 interleaved accs).
// h A-operand staged in LDS in fragment-ready layout (bulk 16B copies), so the
// inner loop feeds WMMA from ds_load instead of stalling on global loadcnt.
// -------------------------------------------------------------------------
__global__ __launch_bounds__(256)
void lstm_step_kernel(const _Float16* __restrict__ xt,   // [B][DIN] f16
                      const _Float16* __restrict__ hin,  // [B][HID] f16
                      _Float16* __restrict__ hout,       // [B][HID] f16
                      float* __restrict__ cst,           // [B][HID] f32 (in-place)
                      _Float16* __restrict__ hs_out,     // [B][HID] f16 slot at t
                      const _Float16* __restrict__ wih,  // panel [6][128][512]
                      const _Float16* __restrict__ whh,  // panel [16][128][512]
                      const float* __restrict__ bias) {  // [4*HID]
    // 2 local M-tiles x 16 K-groups x 512 halves = 32 KB, fragment-ready layout
    __shared__ __align__(32) _Float16 hpan[2 * 16 * 512];
    float* gls = (float*)hpan;   // 32 rows x 64 cols = 8 KB, reused after barrier

    const int wg    = blockIdx.x;          // 0..63
    const int lane  = threadIdx.x & 31;
    const int wv    = threadIdx.x >> 5;    // 0..7
    const int gate  = wv >> 1;             // 0..3 (i,f,g,o)
    const int mtl   = wv & 1;              // local M-tile 0/1
    const int rbase = (wg & 1) * 32;       // batch-row base of this WG
    const int ntile = gate * 32 + (wg >> 1);  // global 16-col tile index

    const _Float16* wihp = wih + (size_t)ntile * 512 + lane * 16;
    const _Float16* whhp = whh + (size_t)ntile * 512 + lane * 16;
    __builtin_prefetch(whhp, 0, 3);        // -> global_prefetch_b8 (L2 warm)

    // ---- phase 1: stage h rows [rbase, rbase+32) into LDS, A-fragment layout.
    // Contiguous 8-half source blocks map to contiguous LDS 16B blocks:
    //   dst = ((ml>>4)*16 + kg)*512 + lane2*16 + sb
    //   lane2 = (ml&15) + 16*((kk>>3)&1) ; sb = ((kk>>4)<<3)
    for (int ch = threadIdx.x; ch < 32 * 64; ch += blockDim.x) {
        int ml  = ch >> 6;                 // local row 0..31
        int blk = ch & 63;                 // 8-half block within row
        int k0  = blk * 8;
        int kg  = k0 >> 5;
        int kk  = k0 & 31;
        int lane2 = (ml & 15) + (((kk >> 3) & 1) << 4);
        int sb    = (kk >> 4) << 3;
        int dst   = ((ml >> 4) * 16 + kg) * 512 + lane2 * 16 + sb;
        *(v8h*)(hpan + dst) = *(const v8h*)(hin + (size_t)(rbase + ml) * HID_ + k0);
    }
    __syncthreads();

    // ---- phase 2: one 16x16 D tile per wave, K = 192 (x) + 512 (h) ----
    const int mrow  = rbase + mtl * 16 + (lane & 15);  // global A row
    const int khalf = ((lane >> 4) & 1) * 8;           // lanes 16-31: K+8

    v8f zero = {0.f, 0.f, 0.f, 0.f, 0.f, 0.f, 0.f, 0.f};
    v8f accA = zero, accB = zero;   // two interleaved accumulators

    // input projection: K = 192 (6 groups of 32), A direct from global (L2)
#pragma unroll
    for (int kg = 0; kg < DIN_ / 32; ++kg) {
        v16h bv = *(const v16h*)(wihp + (size_t)kg * (G4_ / 16) * 512);
        v16h av = load_afrag(xt, mrow, DIN_, kg * 32 + khalf);
        if (kg & 1) accB = wmma_f16(av, bv, accB);
        else        accA = wmma_f16(av, bv, accA);
    }
    // recurrent projection: K = 512 (16 groups of 32), A from LDS panel
#pragma unroll
    for (int kg = 0; kg < HID_ / 32; ++kg) {
        v16h bv = *(const v16h*)(whhp + (size_t)kg * (G4_ / 16) * 512);
        v16h av = *(const v16h*)(hpan + ((mtl * 16 + kg) * 512 + lane * 16));
        if (kg & 1) accB = wmma_f16(av, bv, accB);
        else        accA = wmma_f16(av, bv, accA);
    }
    v8f acc = accA + accB;

    // ---- phase 3: scatter D tile to LDS gate buffer (reuses hpan) ----
    __syncthreads();   // all hpan reads done before overwrite
    // D layout: VGPR r, lanes 0-15 -> M=r, lanes 16-31 -> M=r+8; N = lane&15.
    const int nloc = lane & 15;
    const int mhi  = (lane >> 4) << 3;            // 0 or 8
    const int col  = gate * 16 + nloc;            // 0..63
#pragma unroll
    for (int r = 0; r < 8; ++r) {
        gls[(mtl * 16 + mhi + r) * 64 + col] = acc[r];
    }
    __syncthreads();

    // ---- phase 4: gates (torch order i,f,g,o), update c & h ----
    for (int e = threadIdx.x; e < 32 * 16; e += blockDim.x) {
        int bbl = e >> 4;                  // local row 0..31
        int jj  = e & 15;
        int b   = rbase + bbl;
        int j   = (wg >> 1) * 16 + jj;
        float gi = gls[bbl * 64 +  0 + jj] + bias[0 * HID_ + j];
        float gf = gls[bbl * 64 + 16 + jj] + bias[1 * HID_ + j];
        float gg = gls[bbl * 64 + 32 + jj] + bias[2 * HID_ + j];
        float go = gls[bbl * 64 + 48 + jj] + bias[3 * HID_ + j];
        float cp = cst[(size_t)b * HID_ + j];
        float si = 1.0f / (1.0f + __expf(-gi));
        float sf = 1.0f / (1.0f + __expf(-gf));
        float so = 1.0f / (1.0f + __expf(-go));
        float cn = sf * cp + si * tanhf(gg);
        float hn = so * tanhf(cn);
        cst[(size_t)b * HID_ + j] = cn;
        _Float16 hh = (_Float16)hn;
        hout[(size_t)b * HID_ + j]   = hh;
        hs_out[(size_t)b * HID_ + j] = hh;
    }
}

// -------------------------------------------------------------------------
// emission[b][t][c] = [hs_f ; hs_b] @ w_cls^T + b_cls
// One thread per (b,t): reads each hs element exactly once, 10 class accs.
// -------------------------------------------------------------------------
__global__ void emission_kernel(const _Float16* __restrict__ hsf,
                                const _Float16* __restrict__ hsb,
                                const float* __restrict__ wcls,
                                const float* __restrict__ bcls,
                                float* __restrict__ feats) {
    int total = B_ * T_;
    for (int e = blockIdx.x * blockDim.x + threadIdx.x; e < total;
         e += gridDim.x * blockDim.x) {
        int t = e % T_;
        int b = e / T_;
        const _Float16* hf = hsf + (size_t)(t * B_ + b) * HID_;
        const _Float16* hb = hsb + (size_t)(t * B_ + b) * HID_;
        float acc[CLS_];
#pragma unroll
        for (int cc = 0; cc < CLS_; ++cc) acc[cc] = bcls[cc];
        for (int j = 0; j < HID_; ++j) {
            float xf = (float)hf[j];
            float xb = (float)hb[j];
#pragma unroll
            for (int cc = 0; cc < CLS_; ++cc) {
                acc[cc] += xf * wcls[cc * (2 * HID_) + j]
                         + xb * wcls[cc * (2 * HID_) + HID_ + j];
            }
        }
        float* out = feats + (size_t)(b * T_ + t) * CLS_;
#pragma unroll
        for (int cc = 0; cc < CLS_; ++cc) out[cc] = acc[cc];
    }
}

// -------------------------------------------------------------------------
// CRF partition function (one lane per batch element)
// -------------------------------------------------------------------------
__global__ void crf_forward_kernel(const float* __restrict__ feats,
                                   const float* __restrict__ trans,
                                   float* __restrict__ out) {
    int b = threadIdx.x;
    if (b >= B_) return;
    float alpha[CLS_];
#pragma unroll
    for (int c = 0; c < CLS_; ++c) alpha[c] = (c == START_ID_) ? 0.0f : NEG_;
    for (int t = 0; t < T_; ++t) {
        const float* ft = feats + (size_t)(b * T_ + t) * CLS_;
        float na[CLS_];
#pragma unroll
        for (int to = 0; to < CLS_; ++to) {
            float m = -3.4e38f;
#pragma unroll
            for (int fr = 0; fr < CLS_; ++fr) {
                float v = alpha[fr] + trans[to * CLS_ + fr];
                m = fmaxf(m, v);
            }
            float s = 0.0f;
#pragma unroll
            for (int fr = 0; fr < CLS_; ++fr) {
                float v = alpha[fr] + trans[to * CLS_ + fr];
                s += __expf(v - m);
            }
            na[to] = m + __logf(s) + ft[to];
        }
#pragma unroll
        for (int c = 0; c < CLS_; ++c) alpha[c] = na[c];
    }
    float m = -3.4e38f;
#pragma unroll
    for (int c = 0; c < CLS_; ++c) m = fmaxf(m, alpha[c]);
    float s = 0.0f;
#pragma unroll
    for (int c = 0; c < CLS_; ++c) s += __expf(alpha[c] - m);
    out[b] = m + __logf(s);
}

__global__ void gold_kernel(const float* __restrict__ feats,
                            const float* __restrict__ trans,
                            const int* __restrict__ tags,
                            float* __restrict__ out) {
    int b = threadIdx.x;
    if (b >= B_) return;
    int prev = START_ID_;
    float s = 0.0f;
    for (int t = 0; t < T_; ++t) {
        int tg = tags[b * T_ + t];
        s += trans[tg * CLS_ + prev] + feats[(size_t)(b * T_ + t) * CLS_ + tg];
        prev = tg;
    }
    out[b] = s;
}

__global__ void finalize_kernel(const float* __restrict__ fwd,
                                const float* __restrict__ gold,
                                float* __restrict__ out) {
    __shared__ float red[B_];
    int i = threadIdx.x;
    red[i] = fwd[i] - gold[i];
    __syncthreads();
    for (int off = B_ / 2; off > 0; off >>= 1) {
        if (i < off) red[i] += red[i + off];
        __syncthreads();
    }
    if (i == 0) out[0] = red[0] / (float)B_;
}

// -------------------------------------------------------------------------
// host launcher
// -------------------------------------------------------------------------
extern "C" void kernel_launch(void* const* d_in, const int* in_sizes, int n_in,
                              void* d_out, int out_size, void* d_ws, size_t ws_size,
                              hipStream_t stream) {
    const int*   batch_data  = (const int*)d_in[0];
    const int*   batch_rad   = (const int*)d_in[1];
    const int*   batch_tag   = (const int*)d_in[2];
    const float* char_emb    = (const float*)d_in[3];
    const float* rad_emb     = (const float*)d_in[4];
    const float* w_ih_f      = (const float*)d_in[5];
    const float* w_hh_f      = (const float*)d_in[6];
    const float* b_f         = (const float*)d_in[7];
    const float* w_ih_b      = (const float*)d_in[8];
    const float* w_hh_b      = (const float*)d_in[9];
    const float* b_b         = (const float*)d_in[10];
    const float* w_cls       = (const float*)d_in[11];
    const float* b_cls       = (const float*)d_in[12];
    const float* transition  = (const float*)d_in[13];

    char* ws = (char*)d_ws;
    size_t off = 0;
    auto suballoc = [&](size_t bytes) -> char* {
        char* p = ws + off;
        off = (off + bytes + 255) & ~(size_t)255;
        return p;
    };
    _Float16* X16  = (_Float16*)suballoc((size_t)T_ * B_ * DIN_ * 2);
    _Float16* WIHF = (_Float16*)suballoc((size_t)G4_ * DIN_ * 2);
    _Float16* WIHB = (_Float16*)suballoc((size_t)G4_ * DIN_ * 2);
    _Float16* WHHF = (_Float16*)suballoc((size_t)G4_ * HID_ * 2);
    _Float16* WHHB = (_Float16*)suballoc((size_t)G4_ * HID_ * 2);
    _Float16* HFA  = (_Float16*)suballoc((size_t)B_ * HID_ * 2);
    _Float16* HFB  = (_Float16*)suballoc((size_t)B_ * HID_ * 2);
    _Float16* HBA  = (_Float16*)suballoc((size_t)B_ * HID_ * 2);
    _Float16* HBB  = (_Float16*)suballoc((size_t)B_ * HID_ * 2);
    float*    CF   = (float*)suballoc((size_t)B_ * HID_ * 4);
    float*    CB   = (float*)suballoc((size_t)B_ * HID_ * 4);
    _Float16* HSF  = (_Float16*)suballoc((size_t)T_ * B_ * HID_ * 2);
    _Float16* HSB  = (_Float16*)suballoc((size_t)T_ * B_ * HID_ * 2);
    float*    FEATS= (float*)suballoc((size_t)B_ * T_ * CLS_ * 4);
    float*    FWD  = (float*)suballoc((size_t)B_ * 4);
    float*    GOLD = (float*)suballoc((size_t)B_ * 4);

    // 1) pack weights to f16 WMMA B-panels (L2-resident afterwards)
    pack_weight_kernel<<<(G4_ * DIN_ + 255) / 256, 256, 0, stream>>>(w_ih_f, WIHF, DIN_);
    pack_weight_kernel<<<(G4_ * DIN_ + 255) / 256, 256, 0, stream>>>(w_ih_b, WIHB, DIN_);
    pack_weight_kernel<<<(G4_ * HID_ + 255) / 256, 256, 0, stream>>>(w_hh_f, WHHF, HID_);
    pack_weight_kernel<<<(G4_ * HID_ + 255) / 256, 256, 0, stream>>>(w_hh_b, WHHB, HID_);

    // 2) embedding gather -> f16 inputs
    embed_kernel<<<(T_ * B_ * DIN_ + 255) / 256, 256, 0, stream>>>(
        batch_data, batch_rad, char_emb, rad_emb, X16);

    // 3) zero initial states
    init_state_kernel<<<(B_ * HID_ + 255) / 256, 256, 0, stream>>>(HFA, HBA, CF, CB);

    // 4) bidirectional LSTM recurrence: one WMMA kernel per step per direction
    for (int t = 0; t < T_; ++t) {
        const _Float16* hinF = (t & 1) ? HFB : HFA;
        _Float16*       houtF = (t & 1) ? HFA : HFB;
        lstm_step_kernel<<<64, 256, 0, stream>>>(
            X16 + (size_t)t * B_ * DIN_, hinF, houtF, CF,
            HSF + (size_t)t * B_ * HID_, WIHF, WHHF, b_f);

        int tb = T_ - 1 - t;
        const _Float16* hinB = (t & 1) ? HBB : HBA;
        _Float16*       houtB = (t & 1) ? HBA : HBB;
        lstm_step_kernel<<<64, 256, 0, stream>>>(
            X16 + (size_t)tb * B_ * DIN_, hinB, houtB, CB,
            HSB + (size_t)tb * B_ * HID_, WIHB, WHHB, b_b);
    }

    // 5) emissions
    emission_kernel<<<(B_ * T_ + 255) / 256, 256, 0, stream>>>(
        HSF, HSB, w_cls, b_cls, FEATS);

    // 6) CRF partition, gold score, loss
    crf_forward_kernel<<<1, B_, 0, stream>>>(FEATS, transition, FWD);
    gold_kernel<<<1, B_, 0, stream>>>(FEATS, transition, batch_tag, GOLD);
    finalize_kernel<<<1, B_, 0, stream>>>(FWD, GOLD, (float*)d_out);
}
